// EncoderLayer_34282428957083
// MI455X (gfx1250) — matrix operand order repaired and checked by reading, not compiled, not run on hardware
//
#include <hip/hip_runtime.h>

// ---------------------------------------------------------------------------
// EncoderLayer for MI455X (gfx1250): bf16 WMMA + Tensor Data Mover staging.
// B=4, S=1024, D=1024, H=16, DH=64, DFF=2048
// ---------------------------------------------------------------------------

typedef __bf16 bf16;
typedef __attribute__((ext_vector_type(16))) __bf16 v16bf;
typedef __attribute__((ext_vector_type(8)))  __bf16 v8bf;
typedef __attribute__((ext_vector_type(8)))  float  v8f;
typedef unsigned int u32x4_t __attribute__((ext_vector_type(4)));
typedef int          i32x4_t __attribute__((ext_vector_type(4)));
typedef int          i32x8_t __attribute__((ext_vector_type(8)));

struct __attribute__((aligned(8))) bf16x4 { bf16 v[4]; };

#define BB   4
#define SS   1024
#define DD   1024
#define HH   16
#define DHH  64
#define DFF  2048

// TDM availability (device pass only; host pass falls back so both parse)
#if defined(__has_builtin)
#if __has_builtin(__builtin_amdgcn_tensor_load_to_lds) && \
    __has_builtin(__builtin_amdgcn_s_wait_tensorcnt)
#define USE_TDM 1
#endif
#endif
#ifndef USE_TDM
#define USE_TDM 0
#endif

__device__ __forceinline__ bf16x4 cvt4(const float* src) {
    float4 f = *(const float4*)src;
    bf16x4 r;
    r.v[0] = (bf16)f.x; r.v[1] = (bf16)f.y; r.v[2] = (bf16)f.z; r.v[3] = (bf16)f.w;
    return r;
}

#if USE_TDM
// Issue one 2D TDM tile load: tile_d0 elements (bf16) per row, tile_d1 rows,
// global row stride `stride` (elements). LDS rows are padded: after
// pad_interval_code dwords insert pad_amount_code+1... (codes per ISA:
// interval 3 -> 16 dwords, 4 -> 32 dwords; amount 3 -> 4 dwords).
__device__ __forceinline__ void tdm_load_2d(unsigned int lds_off, const void* gptr,
                                            int tile_d0, int tile_d1,
                                            long long stride,
                                            unsigned int pad_interval_code,
                                            unsigned int pad_amount_code)
{
    unsigned long long ga = (unsigned long long)(uintptr_t)gptr;
    u32x4_t g0;
    g0[0] = 1u;                                   // count=1, user descriptor
    g0[1] = lds_off;                              // lds_addr (bytes)
    g0[2] = (unsigned int)ga;                     // global_addr[31:0]
    g0[3] = ((unsigned int)(ga >> 32) & 0x01FFFFFFu) | (2u << 30); // addr[56:32] | type=2
    i32x8_t g1;
    g1[0] = (int)((1u << 16) |                    // data_size = 1 (2 bytes)
                  (1u << 20) |                    // pad_enable
                  (pad_interval_code << 22) |
                  (pad_amount_code << 25));
    g1[1] = (int)(((unsigned int)tile_d0 & 0xFFFFu) << 16);   // tensor_dim0[15:0]
    g1[2] = (int)(((unsigned int)tile_d1 & 0xFFFFu) << 16);   // tensor_dim1[15:0]
    g1[3] = (int)(((unsigned int)tile_d0 & 0xFFFFu) << 16);   // tile_dim0
    g1[4] = (int)((unsigned int)tile_d1 & 0xFFFFu);           // tile_dim1 (tile_dim2=0)
    g1[5] = (int)(unsigned int)(stride & 0xFFFFFFFFll);       // dim0_stride[31:0]
    g1[6] = (int)(unsigned int)((stride >> 32) & 0xFFFFll);   // dim0_stride[47:32]
    g1[7] = 0;
    i32x4_t z4 = {0, 0, 0, 0};
#if defined(__clang_major__) && (__clang_major__ >= 23)
    i32x8_t z8 = {0, 0, 0, 0, 0, 0, 0, 0};
    __builtin_amdgcn_tensor_load_to_lds(g0, g1, z4, z4, z8, 0);
#else
    __builtin_amdgcn_tensor_load_to_lds(g0, g1, z4, z4, 0);
#endif
}
#endif

// WMMA 16-bit A-fragment (16x32): lanes 0-15 K=0..7/16..23, lanes 16-31
// K=8..15/24..31, row = lane&15.
__device__ __forceinline__ v16bf frag_a(const bf16* rowptr, int hi) {
    v8bf lo = *(const v8bf*)(rowptr + hi * 8);
    v8bf hp = *(const v8bf*)(rowptr + 16 + hi * 8);
    v16bf a;
#pragma unroll
    for (int e = 0; e < 8; ++e) { a[e] = lo[e]; a[8 + e] = hp[e]; }
    return a;
}

// WMMA 16-bit B-fragment (32x16) from transposed storage Bt[n][k]:
// lanes 0-15 K=0..7/16..23, lanes 16-31 K=8..15/24..31, N = lane&15.
__device__ __forceinline__ v16bf frag_b(const bf16* ncolptr, int hi) {
    v8bf lo = *(const v8bf*)(ncolptr + hi * 8);
    v8bf hp = *(const v8bf*)(ncolptr + 16 + hi * 8);
    v16bf b;
#pragma unroll
    for (int e = 0; e < 8; ++e) { b[e] = lo[e]; b[8 + e] = hp[e]; }
    return b;
}

// ---------------------------------------------------------------------------
// Prep kernels: fp32 -> bf16 convert, and fused transpose-convert so every
// GEMM B operand is pre-stored as Bt[N][K] (TDM-friendly, no LDS transpose).
// ---------------------------------------------------------------------------
__global__ __launch_bounds__(256)
void cvt_f32_to_bf16(const float* __restrict__ s, bf16* __restrict__ d, int n4)
{
    int i = blockIdx.x * 256 + threadIdx.x;
    if (i < n4) *(bf16x4*)(d + (size_t)i * 4) = cvt4(s + (size_t)i * 4);
}

__global__ __launch_bounds__(256)
void transpose_cvt(const float* __restrict__ s, bf16* __restrict__ d, int R, int C)
{
    size_t bo = (size_t)blockIdx.y * R * C;
    int i = blockIdx.x * 256 + threadIdx.x;
    if (i < R * C) {
        int r = i / C, c = i % C;
        d[bo + (size_t)c * R + r] = (bf16)s[bo + i];
    }
}

// ---------------------------------------------------------------------------
// GEMM: C[M x N] = act(A[M x K] * Bt[N x K]^T + bias).  128x64 tile per
// block, 128 threads = 4 waves; each wave owns 32 rows -> 8 WMMAs reuse 4
// B-fragments per 32-deep K-step.  Tiles staged by TDM (wave 0) or the
// vectorized cooperative fallback.  vtrans epilogue writes V as [B,H,DH,S].
// ---------------------------------------------------------------------------
__global__ __launch_bounds__(128)
void gemm_bt_wmma(const bf16* __restrict__ A, int lda,
                  const bf16* __restrict__ Bt, int ldbt,
                  const float* __restrict__ bias,
                  float* __restrict__ Cf, bf16* __restrict__ Cbf,
                  int ldc, int coff, int K, int relu, int vtrans)
{
    __shared__ __align__(16) bf16 As[128][40];   // 32 data + 8 pad per row
    __shared__ __align__(16) bf16 Bs[64][40];

    const int tid  = threadIdx.x;
    const int w    = tid >> 5;
    const int lane = tid & 31;
    const int ln   = lane & 15;
    const int hi   = lane >> 4;
    const int m0   = blockIdx.x * 128;
    const int n0   = blockIdx.y * 64;

    v8f zero = {0.f, 0.f, 0.f, 0.f, 0.f, 0.f, 0.f, 0.f};
    v8f acc[2][4];
#pragma unroll
    for (int mA = 0; mA < 2; ++mA)
#pragma unroll
        for (int t = 0; t < 4; ++t) acc[mA][t] = zero;

    for (int k0 = 0; k0 < K; k0 += 32) {
        __syncthreads();
#if USE_TDM
        if (tid < 32) {
            tdm_load_2d((unsigned int)(uintptr_t)&As[0][0],
                        A + (size_t)m0 * lda + k0, 32, 128, lda, 3u, 3u);
            tdm_load_2d((unsigned int)(uintptr_t)&Bs[0][0],
                        Bt + (size_t)n0 * ldbt + k0, 32, 64, ldbt, 3u, 3u);
            __builtin_amdgcn_s_wait_tensorcnt((short)0);
        }
#else
        for (int idx = tid * 4; idx < 128 * 32; idx += 512) {
            int r = idx >> 5, c = idx & 31;
            *(bf16x4*)&As[r][c] = *(const bf16x4*)(A + (size_t)(m0 + r) * lda + k0 + c);
        }
        for (int idx = tid * 4; idx < 64 * 32; idx += 512) {
            int r = idx >> 5, c = idx & 31;
            *(bf16x4*)&Bs[r][c] = *(const bf16x4*)(Bt + (size_t)(n0 + r) * ldbt + k0 + c);
        }
#endif
        __syncthreads();

        v16bf a0 = frag_a(&As[w * 32 + ln][0], hi);
        v16bf a1 = frag_a(&As[w * 32 + 16 + ln][0], hi);
#pragma unroll
        for (int t = 0; t < 4; ++t) {
            v16bf b = frag_b(&Bs[t * 16 + ln][0], hi);
            acc[0][t] = __builtin_amdgcn_wmma_f32_16x16x32_bf16(
                false, a0, false, b, (short)0, acc[0][t], false, false);
            acc[1][t] = __builtin_amdgcn_wmma_f32_16x16x32_bf16(
                false, a1, false, b, (short)0, acc[1][t], false, false);
        }
    }

#pragma unroll
    for (int mA = 0; mA < 2; ++mA) {
#pragma unroll
        for (int t = 0; t < 4; ++t) {
#pragma unroll
            for (int r = 0; r < 8; ++r) {
                int gm = m0 + w * 32 + mA * 16 + r + hi * 8;
                int gn = n0 + t * 16 + ln;
                float vo = acc[mA][t][r];
                if (bias) vo += bias[gn];
                if (relu) vo = fmaxf(vo, 0.0f);
                if (vtrans) {   // write V transposed: [B, H, DH, S]
                    int b = gm >> 10, s = gm & (SS - 1);
                    Cbf[(size_t)b * HH * DHH * SS + (size_t)(coff + gn) * SS + s] = (bf16)vo;
                } else if (Cbf) {
                    Cbf[(size_t)gm * ldc + coff + gn] = (bf16)vo;
                } else {
                    Cf[(size_t)gm * ldc + coff + gn] = vo;
                }
            }
        }
    }
}

// ---------------------------------------------------------------------------
// Flash attention per (b, h, 64-query tile).  q/k bf16 [B,S,H*DH] row-major;
// v pre-transposed bf16 [B,H,DH,S] so all three tiles are plain 2D TDM tiles.
// ---------------------------------------------------------------------------
__global__ __launch_bounds__(128)
void attn_wmma(const bf16* __restrict__ q, const bf16* __restrict__ k,
               const bf16* __restrict__ vT, bf16* __restrict__ h1)
{
    __shared__ __align__(16) bf16 Qs[64][72];    // q rows x dh    (64 data + 8 pad)
    __shared__ __align__(16) bf16 Ks[64][72];    // key rows x dh  (== Bt for QK^T)
    __shared__ __align__(16) bf16 Vt[64][72];    // dh x key       (== Bt for PV)
    __shared__ __align__(16) bf16 Ps[4][16][72]; // per-wave P staging (C->A relayout)

    const int tid  = threadIdx.x;
    const int w    = tid >> 5;
    const int lane = tid & 31;
    const int ln   = lane & 15;
    const int hi   = lane >> 4;
    const int qb   = blockIdx.x * 64;
    const int bh   = blockIdx.y;
    const int b    = bh / HH;
    const int h    = bh % HH;
    const size_t base    = (size_t)b * SS * DD + (size_t)h * DHH;
    const bf16*  vt_base = vT + ((size_t)b * HH + h) * DHH * SS;

    // stage Q strip once
#if USE_TDM
    if (tid < 32) {
        tdm_load_2d((unsigned int)(uintptr_t)&Qs[0][0],
                    q + base + (size_t)qb * DD, 64, 64, DD, 4u, 3u);
        __builtin_amdgcn_s_wait_tensorcnt((short)0);
    }
#else
    for (int idx = tid * 4; idx < 64 * 64; idx += 512) {
        int r = idx >> 6, c = idx & 63;
        *(bf16x4*)&Qs[r][c] = *(const bf16x4*)(q + base + (size_t)(qb + r) * DD + c);
    }
#endif

    v8f zero = {0.f, 0.f, 0.f, 0.f, 0.f, 0.f, 0.f, 0.f};
    v8f o[4];
    float m_r[8], l_r[8];
#pragma unroll
    for (int t = 0; t < 4; ++t) o[t] = zero;
#pragma unroll
    for (int r = 0; r < 8; ++r) { m_r[r] = -1e30f; l_r[r] = 0.f; }

    for (int kb = 0; kb < SS / 64; ++kb) {
        __syncthreads();
#if USE_TDM
        if (tid < 32) {
            tdm_load_2d((unsigned int)(uintptr_t)&Ks[0][0],
                        k + base + (size_t)(kb * 64) * DD, 64, 64, DD, 4u, 3u);
            tdm_load_2d((unsigned int)(uintptr_t)&Vt[0][0],
                        vt_base + kb * 64, 64, 64, SS, 4u, 3u);
            __builtin_amdgcn_s_wait_tensorcnt((short)0);
        }
#else
        for (int idx = tid * 4; idx < 64 * 64; idx += 512) {
            int r = idx >> 6, c = idx & 63;
            *(bf16x4*)&Ks[r][c] =
                *(const bf16x4*)(k + base + (size_t)(kb * 64 + r) * DD + c);
            *(bf16x4*)&Vt[r][c] =
                *(const bf16x4*)(vt_base + (size_t)r * SS + kb * 64 + c);
        }
#endif
        __syncthreads();

        // scores = Q * K^T  (K-dim = dh = 64 -> two 32-deep WMMA steps)
        v8f sc[4];
#pragma unroll
        for (int t = 0; t < 4; ++t) sc[t] = zero;
#pragma unroll
        for (int kk = 0; kk < 64; kk += 32) {
            v16bf a = frag_a(&Qs[w * 16 + ln][kk], hi);
#pragma unroll
            for (int t = 0; t < 4; ++t) {
                v16bf bfr = frag_b(&Ks[t * 16 + ln][kk], hi);
                sc[t] = __builtin_amdgcn_wmma_f32_16x16x32_bf16(
                    false, a, false, bfr, (short)0, sc[t], false, false);
            }
        }
#pragma unroll
        for (int t = 0; t < 4; ++t)
#pragma unroll
            for (int r = 0; r < 8; ++r) sc[t][r] *= 0.125f;  // 1/sqrt(64)

        // online softmax; reductions stay in each 16-lane half (masks 1/2/4/8)
#pragma unroll
        for (int r = 0; r < 8; ++r) {
            float mx = fmaxf(fmaxf(sc[0][r], sc[1][r]), fmaxf(sc[2][r], sc[3][r]));
#pragma unroll
            for (int offm = 8; offm >= 1; offm >>= 1)
                mx = fmaxf(mx, __shfl_xor(mx, offm, 32));
            float mn    = fmaxf(m_r[r], mx);
            float alpha = __expf(m_r[r] - mn);
            m_r[r] = mn;
            float rs = 0.f;
#pragma unroll
            for (int t = 0; t < 4; ++t) {
                float p = __expf(sc[t][r] - mn);
                sc[t][r] = p;
                rs += p;
            }
#pragma unroll
            for (int offm = 8; offm >= 1; offm >>= 1)
                rs += __shfl_xor(rs, offm, 32);
            l_r[r] = l_r[r] * alpha + rs;
            o[0][r] *= alpha; o[1][r] *= alpha; o[2][r] *= alpha; o[3][r] *= alpha;
#pragma unroll
            for (int t = 0; t < 4; ++t)
                Ps[w][r + 8 * hi][t * 16 + ln] = (bf16)sc[t][r];
        }
        __syncthreads();

        // O += P * V   (K-dim = 64 keys -> two 32-deep WMMA steps)
#pragma unroll
        for (int kk = 0; kk < 64; kk += 32) {
            v16bf a = frag_a(&Ps[w][ln][kk], hi);
#pragma unroll
            for (int t = 0; t < 4; ++t) {
                v16bf bfr = frag_b(&Vt[t * 16 + ln][kk], hi);
                o[t] = __builtin_amdgcn_wmma_f32_16x16x32_bf16(
                    false, a, false, bfr, (short)0, o[t], false, false);
            }
        }
    }

#pragma unroll
    for (int t = 0; t < 4; ++t) {
#pragma unroll
        for (int r = 0; r < 8; ++r) {
            float vo = o[t][r] / l_r[r];
            size_t off = base + (size_t)(qb + w * 16 + r + 8 * hi) * DD + t * 16 + ln;
            h1[off] = (bf16)vo;
        }
    }
}

// ---------------------------------------------------------------------------
// Launch: prep (convert/transpose) -> 48 QKV GEMMs -> attention -> FFN1 -> FFN2
// ws (bf16): xb | Wqt | Wkt | Wvt | W1t | W2t | q | k | vT | h1 | t1  (~70 MB)
// ---------------------------------------------------------------------------
extern "C" void kernel_launch(void* const* d_in, const int* in_sizes, int n_in,
                              void* d_out, int out_size, void* d_ws, size_t ws_size,
                              hipStream_t stream)
{
    (void)in_sizes; (void)n_in; (void)out_size; (void)ws_size;

    const float* x  = (const float*)d_in[0];
    const float* Wq = (const float*)d_in[1];
    const float* Wk = (const float*)d_in[2];
    const float* Wv = (const float*)d_in[3];
    const float* W1 = (const float*)d_in[4];
    const float* b1 = (const float*)d_in[5];
    const float* W2 = (const float*)d_in[6];
    const float* b2 = (const float*)d_in[7];
    float* out = (float*)d_out;

    const size_t NTOK = (size_t)BB * SS;     // 4096
    bf16* xb  = (bf16*)d_ws;
    bf16* Wqt = xb  + NTOK * DD;             // [H][DH][D]
    bf16* Wkt = Wqt + (size_t)HH * DHH * DD;
    bf16* Wvt = Wkt + (size_t)HH * DHH * DD;
    bf16* W1t = Wvt + (size_t)HH * DHH * DD; // [DFF][D]
    bf16* W2t = W1t + (size_t)DFF * DD;      // [D][DFF]
    bf16* qb  = W2t + (size_t)DD * DFF;      // [B,S,H*DH]
    bf16* kbuf= qb  + NTOK * DD;
    bf16* vT  = kbuf+ NTOK * DD;             // [B,H,DH,S]
    bf16* h1  = vT  + NTOK * DD;             // [B,S,D]
    bf16* t1  = h1  + NTOK * DD;             // [B*S, DFF]

    dim3 b256(256);
    // prep: convert x; transpose-convert all weights into Bt[N][K] form
    cvt_f32_to_bf16<<<dim3((unsigned)(NTOK * DD / 4 / 256)), b256, 0, stream>>>(
        x, xb, (int)(NTOK * DD / 4));
    transpose_cvt<<<dim3((DD * DHH + 255) / 256, HH), b256, 0, stream>>>(Wq, Wqt, DD, DHH);
    transpose_cvt<<<dim3((DD * DHH + 255) / 256, HH), b256, 0, stream>>>(Wk, Wkt, DD, DHH);
    transpose_cvt<<<dim3((DD * DHH + 255) / 256, HH), b256, 0, stream>>>(Wv, Wvt, DD, DHH);
    transpose_cvt<<<dim3((DD * DFF + 255) / 256, 1), b256, 0, stream>>>(W1, W1t, DD, DFF);
    transpose_cvt<<<dim3((DFF * DD + 255) / 256, 1), b256, 0, stream>>>(W2, W2t, DFF, DD);

    dim3 blk(128);
    // QKV projections: per-head [4096 x 1024] @ [1024 x 64]
    for (int h = 0; h < HH; ++h) {
        dim3 g((unsigned)(NTOK / 128), 1);
        gemm_bt_wmma<<<g, blk, 0, stream>>>(
            xb, DD, Wqt + (size_t)h * DHH * DD, DD, nullptr,
            nullptr, qb, DD, h * DHH, DD, 0, 0);
        gemm_bt_wmma<<<g, blk, 0, stream>>>(
            xb, DD, Wkt + (size_t)h * DHH * DD, DD, nullptr,
            nullptr, kbuf, DD, h * DHH, DD, 0, 0);
        gemm_bt_wmma<<<g, blk, 0, stream>>>(
            xb, DD, Wvt + (size_t)h * DHH * DD, DD, nullptr,
            nullptr, vT, 0, h * DHH, DD, 0, 1);     // transposed V epilogue
    }

    // attention
    attn_wmma<<<dim3(SS / 64, BB * HH), blk, 0, stream>>>(qb, kbuf, vT, h1);

    // FFN1: [4096 x 1024] @ W1t[2048 x 1024]^T + b1, relu -> bf16
    gemm_bt_wmma<<<dim3((unsigned)(NTOK / 128), DFF / 64), blk, 0, stream>>>(
        h1, DD, W1t, DD, b1, nullptr, t1, DFF, 0, DD, 1, 0);

    // FFN2: [4096 x 2048] @ W2t[1024 x 2048]^T + b2 -> f32 out
    gemm_bt_wmma<<<dim3((unsigned)(NTOK / 128), DD / 64), blk, 0, stream>>>(
        t1, DFF, W2t, DFF, b2, out, nullptr, DD, 0, DFF, 0, 0);
}
